// HierDDLtsC_63410897158388
// MI455X (gfx1250) — matrix operand order repaired and checked by reading, not compiled, not run on hardware
//
#include <hip/hip_runtime.h>

// ---------------------------------------------------------------------------
// MI455X (gfx1250) implementation.
//
// Pipeline (fp32 intermediates, bf16 WMMA operands):
//   1. W0[k,i,j] = sum_g P[i,j,g]*cos(2*pi*k/period)      (2048,32,32)  [VALU]
//   2. T0[b,k,i] = sum_j emb[tok[b,k],j]*W0[k,i,j]        (8,2048,32)   [VALU + async LDS]
//   3. h0 = CL^T @ T0  folded over batch:                 GEMM 512x256x2048 [WMMA]
//   4. Z1 = h0 @ M1^T                                     GEMM 4096x64x32  [WMMA]
//   5. LN -> Zn1; W1 precompute (reuse W slab); Tp1 = pos(Zn1)+Z1 [VALU + async LDS]
//   6. h1 = L1^T @ Tp1 (U+res fused: shared Linker)       GEMM 512x512x512 [WMMA]
//   7. Z2 = h1 @ M2^T                                     GEMM 4096x64x64  [WMMA]
//   8. LN -> Zn2; W2; Tp2 = pos(Zn2)+Z2                   [VALU + async LDS]
//   9. out = L2^T @ Tp2                                   GEMM 256x512x512 [WMMA]
//
// GEMM: 2x2 register tiling (32x32 per wave) + software pipelining so loads
// for K-step k+32 are in flight while the 4 WMMAs of step k issue.
// ---------------------------------------------------------------------------

typedef __attribute__((ext_vector_type(16))) __bf16 v16bf;
typedef __attribute__((ext_vector_type(8)))  __bf16 v8bf;
typedef __attribute__((ext_vector_type(8)))  float  v8f;

#define PI2F 6.28318530717958647692f

// ---- async global -> LDS staging (ASYNCcnt-tracked, gfx1250) ----
__device__ __forceinline__ void async_ld_f32_to_lds(const float* gsrc, float* lds_dst) {
  // Generic-pointer low 32 bits == LDS byte offset (ISA 10.2 aperture rule).
  unsigned lds_off = (unsigned)(unsigned long long)lds_dst;
  asm volatile("global_load_async_to_lds_b32 %0, %1, off"
               :: "v"(lds_off), "v"(gsrc) : "memory");
}
__device__ __forceinline__ void wait_async_lds() {
  asm volatile("s_wait_asynccnt 0x0" ::: "memory");
}

// ---- basis precompute: W[k,i,j] = sum_g P[i,j,g] * cos(2*pi*k/period) ----
__global__ void k_precompute_W(const float* __restrict__ P, float* __restrict__ W,
                               int S, int D, int G) {
  int idx = blockIdx.x * blockDim.x + threadIdx.x;
  int total = S * D * D;
  if (idx >= total) return;
  int j = idx % D;
  int i = (idx / D) % D;
  int k = idx / (D * D);
  float kk = (float)k;
  int base = (i * D + j) * G;
  float acc = 0.f;
  for (int g = 0; g < G; ++g) {
    float period = (float)(base + g + 2);
    acc = __builtin_fmaf(P[base + g], cosf(PI2F * kk / period), acc);
  }
  W[idx] = acc;
}

// ---- char layer: embed gather + per-position 32x32 transform ----
__global__ void k_char_pos(const int* __restrict__ tokens, const float* __restrict__ emb,
                           const float* __restrict__ W, float* __restrict__ T0) {
  const int k = blockIdx.x;   // 0..2047
  const int b = blockIdx.y;   // 0..7
  const int i = threadIdx.x;  // 0..31
  __shared__ float z[32];
  int tok = tokens[b * 2048 + k];
  async_ld_f32_to_lds(emb + tok * 32 + i, &z[i]);
  wait_async_lds();
  __syncthreads();
  const float* wrow = W + (size_t)k * 1024 + i * 32;
  float acc = 0.f;
  #pragma unroll
  for (int j = 0; j < 32; ++j) acc = __builtin_fmaf(z[j], wrow[j], acc);
  T0[((size_t)b * 2048 + k) * 32 + i] = acc;
}

// ---- per-position 64x64 transform + shared residual add ----
__global__ void k_pos_residual(const float* __restrict__ Zn, const float* __restrict__ Z,
                               const float* __restrict__ W, float* __restrict__ Tp) {
  const int s = blockIdx.x;   // 0..511
  const int b = blockIdx.y;   // 0..7
  const int i = threadIdx.x;  // 0..63
  __shared__ float z[64];
  size_t row = ((size_t)b * 512 + s) * 64;
  async_ld_f32_to_lds(Zn + row + i, &z[i]);
  wait_async_lds();
  __syncthreads();
  const float* wrow = W + (size_t)s * 4096 + i * 64;
  float acc = 0.f;
  #pragma unroll
  for (int j = 0; j < 64; ++j) acc = __builtin_fmaf(z[j], wrow[j], acc);
  Tp[row + i] = acc + Z[row + i];
}

// ---- layernorm, one row (length D) per thread ----
__global__ void k_layernorm(const float* __restrict__ X, const float* __restrict__ gamma,
                            const float* __restrict__ beta, float* __restrict__ Y,
                            int rows, int D) {
  int r = blockIdx.x * blockDim.x + threadIdx.x;
  if (r >= rows) return;
  const float* x = X + (size_t)r * D;
  float m = 0.f;
  for (int i = 0; i < D; ++i) m += x[i];
  m /= (float)D;
  float v = 0.f;
  for (int i = 0; i < D; ++i) { float d = x[i] - m; v = __builtin_fmaf(d, d, v); }
  v /= (float)D;
  float inv = rsqrtf(v + 1e-5f);
  float* y = Y + (size_t)r * D;
  for (int i = 0; i < D; ++i) y[i] = (x[i] - m) * inv * gamma[i] + beta[i];
}

// ---- layout conversion kernels (fp32 -> bf16 WMMA operand arenas) ----
__global__ void k_f32_to_bf16(const float* __restrict__ in, __bf16* __restrict__ out, int n) {
  int i = blockIdx.x * blockDim.x + threadIdx.x;
  if (i < n) out[i] = (__bf16)in[i];
}

// out[r*C + c] = in[c*R + r]   (in: C x R  ->  out: R x C)
__global__ void k_transpose_bf16(const float* __restrict__ in, __bf16* __restrict__ out,
                                 int R, int C) {
  int idx = blockIdx.x * blockDim.x + threadIdx.x;
  if (idx >= R * C) return;
  int c = idx % C;
  int r = idx / C;
  out[idx] = (__bf16)in[(size_t)c * R + r];
}

// Bt[(b*D+d)*S + s] = T[(b*S+s)*D + d]   (per-batch transpose, stacked along N)
__global__ void k_batch_transpose_bf16(const float* __restrict__ in, __bf16* __restrict__ out,
                                       int Bn, int S, int D) {
  int idx = blockIdx.x * blockDim.x + threadIdx.x;
  if (idx >= Bn * S * D) return;
  int s = idx % S;
  int t = idx / S;
  int d = t % D;
  int b = t / D;
  out[idx] = (__bf16)in[((size_t)b * S + s) * D + d];
}

// out[(b*S+t)*D + i] = in[t*(Bn*D) + b*D + i]   (unfold batch-folded GEMM output)
template <typename T>
__global__ void k_unfold(const float* __restrict__ in, T* __restrict__ out,
                         int Bn, int S, int D) {
  int idx = blockIdx.x * blockDim.x + threadIdx.x;
  if (idx >= Bn * S * D) return;
  int i = idx % D;
  int t = (idx / D) % S;
  int b = idx / (D * S);
  out[idx] = (T)in[(size_t)t * (Bn * D) + b * D + i];
}

// ---- WMMA fragment loads per CDNA5 ISA 7.12.2 VGPR layouts ----
// A frag (16-bit A 16x32): lane m=lane%16, h=lane/16; K chunks [8h,8h+8) & [16+8h,24+8h)
__device__ __forceinline__ v16bf load_a_frag(const __bf16* row, int k, int h) {
  v8bf lo = *(const v8bf*)(row + k + 8 * h);
  v8bf hi = *(const v8bf*)(row + k + 16 + 8 * h);
  v16bf a;
  #pragma unroll
  for (int e = 0; e < 8; ++e) { a[e] = lo[e]; a[e + 8] = hi[e]; }
  return a;
}
// B frag (16-bit B 32x16): lane n=lane%16, h=lane/16; K chunks [8h,8h+8) & [16+8h,24+8h)
__device__ __forceinline__ v16bf load_b_frag(const __bf16* row, int k, int h) {
  v8bf lo = *(const v8bf*)(row + k + 8 * h);
  v8bf hi = *(const v8bf*)(row + k + 16 + 8 * h);
  v16bf b;
  #pragma unroll
  for (int e = 0; e < 8; ++e) { b[e] = lo[e]; b[e + 8] = hi[e]; }
  return b;
}
__device__ __forceinline__ v8f wmma_bf16(v16bf a, v16bf b, v8f c) {
  return __builtin_amdgcn_wmma_f32_16x16x32_bf16(false, a, false, b, (short)0, c, false, false);
}

// ---- bf16 WMMA GEMM: C[MxN] = A[MxK] * Bt[NxK]^T, fp32 accumulate ----
// Wave computes 32x32 (2x2 tiles of 16x16); 4 waves/block as 2x2 -> 64x64 block.
// Software-pipelined: next K-step fragments load while current WMMAs issue.
__global__ void k_wmma_gemm_bf16(const __bf16* __restrict__ A, const __bf16* __restrict__ Bt,
                                 float* __restrict__ C, int M, int N, int K) {
  const int lane = threadIdx.x & 31;
  const int r = lane & 15;
  const int h = lane >> 4;
  const int wave = threadIdx.y;            // 0..3
  const int m0 = blockIdx.y * 64 + (wave & 1) * 32;
  const int n0 = blockIdx.x * 64 + (wave >> 1) * 32;
  const __bf16* a0row = A + (size_t)(m0 + r) * K;
  const __bf16* a1row = A + (size_t)(m0 + 16 + r) * K;
  const __bf16* b0row = Bt + (size_t)(n0 + r) * K;
  const __bf16* b1row = Bt + (size_t)(n0 + 16 + r) * K;
  v8f c00 = {}, c01 = {}, c10 = {}, c11 = {};

  v16bf a0 = load_a_frag(a0row, 0, h);
  v16bf a1 = load_a_frag(a1row, 0, h);
  v16bf b0 = load_b_frag(b0row, 0, h);
  v16bf b1 = load_b_frag(b1row, 0, h);
  for (int k = 32; k < K; k += 32) {
    // prefetch one K-slab beyond the in-flight loads
    __builtin_prefetch(a0row + k + 32, 0, 3);
    __builtin_prefetch(b0row + k + 32, 0, 3);
    v16bf na0 = load_a_frag(a0row, k, h);
    v16bf na1 = load_a_frag(a1row, k, h);
    v16bf nb0 = load_b_frag(b0row, k, h);
    v16bf nb1 = load_b_frag(b1row, k, h);
    c00 = wmma_bf16(a0, b0, c00);
    c01 = wmma_bf16(a0, b1, c01);
    c10 = wmma_bf16(a1, b0, c10);
    c11 = wmma_bf16(a1, b1, c11);
    a0 = na0; a1 = na1; b0 = nb0; b1 = nb1;
  }
  c00 = wmma_bf16(a0, b0, c00);
  c01 = wmma_bf16(a0, b1, c01);
  c10 = wmma_bf16(a1, b0, c10);
  c11 = wmma_bf16(a1, b1, c11);

  // C/D frag: lane n=lane%16, VGPR v holds m = v + 8h
  #pragma unroll
  for (int v = 0; v < 8; ++v) {
    int mlo = m0 + v + 8 * h;
    C[(size_t)mlo * N + n0 + r]           = c00[v];
    C[(size_t)mlo * N + n0 + 16 + r]      = c01[v];
    C[(size_t)(mlo + 16) * N + n0 + r]    = c10[v];
    C[(size_t)(mlo + 16) * N + n0 + 16 + r] = c11[v];
  }
}

// ---------------------------------------------------------------------------

static inline size_t align256(size_t x) { return (x + 255) & ~(size_t)255; }
static inline int grid1d(int n) { return (n + 255) / 256; }

extern "C" void kernel_launch(void* const* d_in, const int* in_sizes, int n_in,
                              void* d_out, int out_size, void* d_ws, size_t ws_size,
                              hipStream_t stream) {
  (void)in_sizes; (void)n_in; (void)out_size; (void)ws_size;
  // inputs per setup_inputs() order
  const int*   tokens = (const int*)  d_in[0];   // (8,2048)
  const float* emb    = (const float*)d_in[1];   // (16,32)
  const float* charP  = (const float*)d_in[2];   // (32,32,8)
  const float* CL     = (const float*)d_in[3];   // (2048,512)
  const float* M1     = (const float*)d_in[4];   // (64,32)
  const float* P1     = (const float*)d_in[5];   // (64,64,8)
  const float* L1     = (const float*)d_in[6];   // (512,512)
  const float* g1     = (const float*)d_in[7];
  const float* b1     = (const float*)d_in[8];
  const float* M2     = (const float*)d_in[9];   // (64,64)
  const float* P2     = (const float*)d_in[10];  // (64,64,8)
  const float* L2     = (const float*)d_in[11];  // (512,256)
  const float* g2     = (const float*)d_in[12];
  const float* b2     = (const float*)d_in[13];
  float* out = (float*)d_out;                    // (8,256,64)

  // ---- bump-allocate scratch arena (~25 MB) ----
  char* ws = (char*)d_ws;
  size_t off = 0;
  auto alloc = [&](size_t bytes) -> void* {
    void* p = ws + off;
    off = align256(off + bytes);
    return p;
  };
  float*  W   = (float*) alloc(2097152u * 4);  // reused for W0, W1, W2 (all 2M floats)
  float*  T0  = (float*) alloc(524288u  * 4);  // (8,2048,32)
  float*  C1  = (float*) alloc(131072u  * 4);  // (512,256)  h0 folded
  float*  Z1  = (float*) alloc(262144u  * 4);  // (4096,64)
  float*  Zn1 = (float*) alloc(262144u  * 4);
  float*  Tp1 = (float*) alloc(262144u  * 4);
  float*  C3  = (float*) alloc(262144u  * 4);  // (512,512)  h1 folded
  float*  Z2  = (float*) alloc(262144u  * 4);
  float*  Zn2 = (float*) alloc(262144u  * 4);
  float*  Tp2 = (float*) alloc(262144u  * 4);
  float*  C5  = (float*) alloc(131072u  * 4);  // (256,512)  out folded
  __bf16* A1  = (__bf16*)alloc(1048576u * 2);  // CL^T   (512x2048)
  __bf16* Bt1 = (__bf16*)alloc(524288u  * 2);  // T0 N-major (256x2048)
  __bf16* A2  = (__bf16*)alloc(131072u  * 2);  // h0 rows (4096x32)
  __bf16* Bt2 = (__bf16*)alloc(2048u    * 2);  // M1 (64x32, already N-major)
  __bf16* A3  = (__bf16*)alloc(262144u  * 2);  // L1^T (512x512)
  __bf16* Bt3 = (__bf16*)alloc(262144u  * 2);  // Tp1 N-major (512x512)
  __bf16* A4  = (__bf16*)alloc(262144u  * 2);  // h1 rows (4096x64)
  __bf16* Bt4 = (__bf16*)alloc(4096u    * 2);  // M2 (64x64)
  __bf16* A5  = (__bf16*)alloc(131072u  * 2);  // L2^T (256x512)
  __bf16* Bt5 = (__bf16*)alloc(262144u  * 2);  // Tp2 N-major (512x512)

  const dim3 gblk(32, 4);  // 4 waves/block as 2x2 -> 64x64 output tile

  // ---- char layer ----
  k_precompute_W<<<grid1d(2048 * 32 * 32), 256, 0, stream>>>(charP, W, 2048, 32, 8);
  k_char_pos<<<dim3(2048, 8), 32, 0, stream>>>(tokens, emb, W, T0);
  k_transpose_bf16<<<grid1d(512 * 2048), 256, 0, stream>>>(CL, A1, 512, 2048);
  k_batch_transpose_bf16<<<grid1d(8 * 2048 * 32), 256, 0, stream>>>(T0, Bt1, 8, 2048, 32);
  k_wmma_gemm_bf16<<<dim3(256 / 64, 512 / 64), gblk, 0, stream>>>(A1, Bt1, C1, 512, 256, 2048);

  // ---- hier layer 1 ----
  k_unfold<__bf16><<<grid1d(8 * 512 * 32), 256, 0, stream>>>(C1, A2, 8, 512, 32);
  k_f32_to_bf16<<<grid1d(64 * 32), 256, 0, stream>>>(M1, Bt2, 64 * 32);
  k_wmma_gemm_bf16<<<dim3(64 / 64, 4096 / 64), gblk, 0, stream>>>(A2, Bt2, Z1, 4096, 64, 32);
  k_layernorm<<<grid1d(4096), 256, 0, stream>>>(Z1, g1, b1, Zn1, 4096, 64);
  k_precompute_W<<<grid1d(512 * 64 * 64), 256, 0, stream>>>(P1, W, 512, 64, 8);
  k_pos_residual<<<dim3(512, 8), 64, 0, stream>>>(Zn1, Z1, W, Tp1);
  k_transpose_bf16<<<grid1d(512 * 512), 256, 0, stream>>>(L1, A3, 512, 512);
  k_batch_transpose_bf16<<<grid1d(8 * 512 * 64), 256, 0, stream>>>(Tp1, Bt3, 8, 512, 64);
  k_wmma_gemm_bf16<<<dim3(512 / 64, 512 / 64), gblk, 0, stream>>>(A3, Bt3, C3, 512, 512, 512);

  // ---- hier layer 2 ----
  k_unfold<__bf16><<<grid1d(8 * 512 * 64), 256, 0, stream>>>(C3, A4, 8, 512, 64);
  k_f32_to_bf16<<<grid1d(64 * 64), 256, 0, stream>>>(M2, Bt4, 64 * 64);
  k_wmma_gemm_bf16<<<dim3(64 / 64, 4096 / 64), gblk, 0, stream>>>(A4, Bt4, Z2, 4096, 64, 64);
  k_layernorm<<<grid1d(4096), 256, 0, stream>>>(Z2, g2, b2, Zn2, 4096, 64);
  k_precompute_W<<<grid1d(512 * 64 * 64), 256, 0, stream>>>(P2, W, 512, 64, 8);
  k_pos_residual<<<dim3(512, 8), 64, 0, stream>>>(Zn2, Z2, W, Tp2);
  k_transpose_bf16<<<grid1d(256 * 512), 256, 0, stream>>>(L2, A5, 256, 512);
  k_batch_transpose_bf16<<<grid1d(8 * 512 * 64), 256, 0, stream>>>(Tp2, Bt5, 8, 512, 64);
  k_wmma_gemm_bf16<<<dim3(512 / 64, 256 / 64), gblk, 0, stream>>>(A5, Bt5, C5, 256, 512, 512);

  // ---- unfold final output to (8,256,64) ----
  k_unfold<float><<<grid1d(8 * 256 * 64), 256, 0, stream>>>(C5, out, 8, 256, 64);
}
